// LSHAttention_22282290332151
// MI455X (gfx1250) — matrix operand order repaired and checked by reading, not compiled, not run on hardware
//
#include <hip/hip_runtime.h>
#include <hip/hip_bf16.h>

// ---------------------------------------------------------------------------
// LSH attention (Reformer) for gfx1250.  B=16, L=2048, D=128, R=4, 32 buckets.
// Pipeline:
//   k0: normalize rand_matrix columns
//   k1: normalize q, bf16-cast q/v, LSH hash (argmax of [h,-h])
//   k2: per (b,r) stable bitonic sort of (hash<<11|idx) in LDS
//   k2b: pack per-token round-chunk ids (4x4 bits) for dup-key counting
//   k3: per (b,r,chunk): async global->LDS tile gather, WMMA bf16 S=Q.K^T,
//       mask + (-log count), row m/l, WMMA bf16 O=P.V -> scratch
//   k4: cross-round log-sum-exp combine -> output (exact joint softmax)
// ---------------------------------------------------------------------------

typedef __attribute__((ext_vector_type(16))) __bf16 v16bf;
typedef __attribute__((ext_vector_type(8)))  float  v8f;

constexpr int kBATCH = 2, kHEAD = 8, kLEN = 2048, kD = 128;
constexpr int kR = 4, kNB2 = 16;              // 32 buckets -> 16 chunk pairs
constexpr int kB = kBATCH * kHEAD;            // 16
constexpr float kSCALE    = 0.08838834764831843f;   // 1/sqrt(128)
constexpr float kNEG_BIG  = -1.0e9f;
constexpr float kNEG_SMALL= -1.0e5f;

// ---- workspace layout (bytes) ----
constexpr size_t SZ_FQB = (size_t)kB * kLEN * kD * 2;       // bf16 normalized q
constexpr size_t SZ_IRL = (size_t)kB * kR * kLEN * 4;       // int per (b,r,pos)
constexpr size_t OFF_FQB = 0;
constexpr size_t OFF_VB  = OFF_FQB + SZ_FQB;                // bf16 values
constexpr size_t OFF_RMN = OFF_VB  + SZ_FQB;                // f32 [b][r][k][d]
constexpr size_t SZ_RMN  = (size_t)kB * kR * kNB2 * kD * 4;
constexpr size_t OFF_HSH = OFF_RMN + SZ_RMN;                // int hashes
constexpr size_t OFF_HI  = OFF_HSH + SZ_IRL;                // sorted->orig idx
constexpr size_t OFF_SH  = OFF_HI  + SZ_IRL;                // sorted hashes
constexpr size_t OFF_OI  = OFF_SH  + SZ_IRL;                // orig->sorted pos
constexpr size_t OFF_CHK = OFF_OI  + SZ_IRL;                // packed chunk ids
constexpr size_t SZ_CHK  = (size_t)kB * kLEN * 4;
constexpr size_t OFF_M   = OFF_CHK + SZ_CHK;                // row max
constexpr size_t OFF_L   = OFF_M   + SZ_IRL;                // row sumexp
constexpr size_t OFF_OG  = OFF_L   + SZ_IRL;                // f32 [b][r][L][D]

// ---- LDS layout for k3 (dynamic shared, 312832 B <= 320 KB) ----
constexpr int QS_STR = 136, KS_STR = 136, VT_STR = 264, SS_STR = 264, PP_STR = 528;
constexpr unsigned SM_QS = 0;
constexpr unsigned SM_KS = SM_QS + 128u * QS_STR * 2;   //  34816
constexpr unsigned SM_VT = SM_KS + 256u * KS_STR * 2;   // 104448
constexpr unsigned SM_SS = SM_VT + 128u * VT_STR * 2;   // 172032
constexpr unsigned SM_MT = SM_SS + 128u * SS_STR * 4;   // 307200
constexpr unsigned SMEM3 = SM_MT + 256 * 4 * 4 + 128 * 3 * 4;  // 312832

struct B32x2 { uint4 a, b; };
static_assert(sizeof(B32x2) == sizeof(v16bf), "frag size");

__device__ __forceinline__ v16bf ldfrag(const __bf16* p0, const __bf16* p1) {
  B32x2 t;
  t.a = *(const uint4*)p0;   // ds_load_b128
  t.b = *(const uint4*)p1;
  return __builtin_bit_cast(v16bf, t);
}

// CDNA5 async global->LDS copy, 16 bytes per lane (ASYNCcnt-tracked).
__device__ __forceinline__ void async_g2l_b128(unsigned lds_off, const void* gaddr) {
  asm volatile("global_load_async_to_lds_b128 %0, %1, off"
               :: "v"(lds_off), "v"((unsigned long long)(uintptr_t)gaddr)
               : "memory");
}
__device__ __forceinline__ void wait_async0() {
  asm volatile("s_wait_asynccnt 0" ::: "memory");
}

// ---------------------------------------------------------------- kernel 0 --
__global__ void lsh_rmnorm(const float* __restrict__ rm, float* __restrict__ rmn) {
  const int k = blockIdx.x, r = blockIdx.y, b = blockIdx.z, d = threadIdx.x;
  __shared__ float red[128];
  const float x = rm[(((size_t)b * kD + d) * kR + r) * kNB2 + k];
  red[d] = x * x; __syncthreads();
  for (int s = 64; s > 0; s >>= 1) { if (d < s) red[d] += red[d + s]; __syncthreads(); }
  const float rn = rsqrtf(red[0] + 1e-30f);
  rmn[(((size_t)b * kR + r) * kNB2 + k) * kD + d] = x * rn;
}

// ---------------------------------------------------------------- kernel 1 --
__global__ void lsh_hash(const float* __restrict__ q, const float* __restrict__ v,
                         const float* __restrict__ rmn,
                         __bf16* __restrict__ fqb, __bf16* __restrict__ vbo,
                         int* __restrict__ hashes) {
  const int t = blockIdx.x, b = blockIdx.y, d = threadIdx.x;
  __shared__ float qs[128], red[128], hb[64];
  const size_t row = ((size_t)b * kLEN + t) * kD;
  const float x = q[row + d];
  qs[d] = x; red[d] = x * x; __syncthreads();
  for (int s = 64; s > 0; s >>= 1) { if (d < s) red[d] += red[d + s]; __syncthreads(); }
  const float rn = rsqrtf(red[0] + 1e-30f);
  fqb[row + d] = (__bf16)(x * rn);
  vbo[row + d] = (__bf16)v[row + d];
  if (d < 64) {
    const int r = d >> 4, k = d & 15;
    const float* w = rmn + (((size_t)b * kR + r) * kNB2 + k) * kD;
    float acc = 0.f;
    for (int e = 0; e < kD; ++e) acc += qs[e] * w[e];
    hb[d] = acc;                                   // argmax invariant to rn>0
  }
  __syncthreads();
  if (d < kR) {
    float best = -3.0e38f; int arg = 0;
    for (int k = 0; k < 16; ++k) { float h =  hb[d * 16 + k]; if (h > best) { best = h; arg = k;      } }
    for (int k = 0; k < 16; ++k) { float h = -hb[d * 16 + k]; if (h > best) { best = h; arg = k + 16; } }
    hashes[((size_t)b * kR + d) * kLEN + t] = arg;
  }
}

// ---------------------------------------------------------------- kernel 2 --
__global__ void __launch_bounds__(1024, 1)
lsh_sort(const int* __restrict__ hashes, int* __restrict__ hi,
         int* __restrict__ sh, int* __restrict__ oi) {
  const int r = blockIdx.x, b = blockIdx.y, t = threadIdx.x;
  __shared__ unsigned key[2048];
  const size_t base = ((size_t)b * kR + r) * kLEN;
  key[t]        = ((unsigned)hashes[base + t]        << 11) | (unsigned)t;
  key[t + 1024] = ((unsigned)hashes[base + t + 1024] << 11) | (unsigned)(t + 1024);
  __syncthreads();
  for (unsigned k = 2; k <= 2048; k <<= 1)
    for (unsigned j = k >> 1; j > 0; j >>= 1) {
#pragma unroll
      for (int h = 0; h < 2; ++h) {
        const unsigned i = (unsigned)t + h * 1024, ij = i ^ j;
        if (ij > i) {
          const unsigned a = key[i], c = key[ij];
          const bool up = ((i & k) == 0);
          if ((a > c) == up) { key[i] = c; key[ij] = a; }
        }
      }
      __syncthreads();
    }
#pragma unroll
  for (int h = 0; h < 2; ++h) {
    const unsigned p = (unsigned)t + h * 1024, kk = key[p];
    const unsigned idx = kk & 2047u;
    hi[base + p] = (int)idx; sh[base + p] = (int)(kk >> 11); oi[base + idx] = (int)p;
  }
}

// --------------------------------------------------------------- kernel 2b --
__global__ void lsh_chunks(const int* __restrict__ oi, unsigned* __restrict__ chk) {
  const int g = blockIdx.x * 256 + threadIdx.x;
  const int b = g >> 11, t = g & 2047;
  unsigned w = 0;
#pragma unroll
  for (int r = 0; r < kR; ++r) {
    const unsigned p = (unsigned)oi[((size_t)b * kR + r) * kLEN + t];
    w |= ((p >> 7) & 15u) << (4 * r);
  }
  chk[g] = w;
}

// ---------------------------------------------------------------- kernel 3 --
__global__ void __launch_bounds__(256, 1)
lsh_attn(const __bf16* __restrict__ fqb, const __bf16* __restrict__ vb,
         const int* __restrict__ hi, const int* __restrict__ sh,
         const unsigned* __restrict__ chk, const unsigned char* __restrict__ maskb,
         float* __restrict__ Mg, float* __restrict__ Lg, float* __restrict__ Og) {
  extern __shared__ char smem[];
  __bf16* QS = (__bf16*)(smem + SM_QS);     // [128][136]  queries (bf16)
  __bf16* KS = (__bf16*)(smem + SM_KS);     // [256][136]  window keys (bf16)
  __bf16* VT = (__bf16*)(smem + SM_VT);     // [128][264]  V transposed [d][key]
  float*  SS = (float*) (smem + SM_SS);     // [128][264]  scores (f32)
  __bf16* PP = (__bf16*)(smem + SM_SS);     // same region: P (bf16, stride 528)
  unsigned* kid = (unsigned*)(smem + SM_MT);
  unsigned* khs = kid + 256; unsigned* kck = khs + 256; unsigned* kms = kck + 256;
  unsigned* qid = kms + 256; unsigned* qhs = qid + 128; unsigned* qck = qhs + 128;

  const int n = blockIdx.x, r = blockIdx.y, b = blockIdx.z, tid = threadIdx.x;
  const size_t base = ((size_t)b * kR + r) * kLEN;
  const int np = (n + kNB2 - 1) & (kNB2 - 1);
  const int batch = b / kHEAD;

  { // metadata: 256 key slots (prev chunk ++ cur chunk), 128 query slots
    const int j = tid;
    const int pos = (j < 128) ? (np * 128 + j) : (n * 128 + (j - 128));
    const unsigned id = (unsigned)hi[base + pos];
    kid[j] = id; khs[j] = (unsigned)sh[base + pos];
    kck[j] = chk[(size_t)b * kLEN + id];
    kms[j] = (unsigned)maskb[(size_t)batch * kLEN + id];
    if (tid < 128) {
      const int p2 = n * 128 + tid;
      const unsigned id2 = (unsigned)hi[base + p2];
      qid[tid] = id2; qhs[tid] = (unsigned)sh[base + p2];
      qck[tid] = chk[(size_t)b * kLEN + id2];
    }
  }
  __syncthreads();
  // Q / K tile gather: async global->LDS (16 B per lane, ASYNCcnt-tracked,
  // no VGPR round-trip).  V is gathered transposed via the regular path.
  for (int c = tid; c < 128 * 16; c += 256) {
    const int i = c >> 4, s = c & 15;
    async_g2l_b128(SM_QS + (unsigned)(i * QS_STR + s * 8) * 2,
                   fqb + ((size_t)b * kLEN + qid[i]) * kD + s * 8);
  }
  for (int c = tid; c < 256 * 16; c += 256) {
    const int j = c >> 4, s = c & 15;
    async_g2l_b128(SM_KS + (unsigned)(j * KS_STR + s * 8) * 2,
                   fqb + ((size_t)b * kLEN + kid[j]) * kD + s * 8);
  }
  for (int c = tid; c < 256 * 128; c += 256) {
    const int j = c >> 7, d = c & 127;
    VT[d * VT_STR + j] = vb[((size_t)b * kLEN + kid[j]) * kD + d];
  }
  wait_async0();
  __syncthreads();

  const int w = tid >> 5, lane = tid & 31, lr = lane & 15, lh = lane >> 4;
  const int i0 = w * 16;                     // wave's 16-row strip

  // ---- S = Q . K^T  (bf16 WMMA, K-dim 128 = 4x32) ----
  for (int nt = 0; nt < 16; ++nt) {
    v8f acc = {0.f, 0.f, 0.f, 0.f, 0.f, 0.f, 0.f, 0.f};
#pragma unroll
    for (int kb = 0; kb < 4; ++kb) {
      const int k0 = kb * 32;
      const __bf16* qrow = QS + (i0 + lr) * QS_STR;
      const v16bf A = ldfrag(qrow + k0 + lh * 8, qrow + k0 + 16 + lh * 8);
      const __bf16* krow = KS + (nt * 16 + lr) * KS_STR;
      const v16bf B = ldfrag(krow + k0 + lh * 8, krow + k0 + 16 + lh * 8);
      acc = __builtin_amdgcn_wmma_f32_16x16x32_bf16(false, A, false, B,
                                                    (short)0, acc, false, false);
    }
#pragma unroll
    for (int v = 0; v < 8; ++v)
      SS[(i0 + v + lh * 8) * SS_STR + nt * 16 + lr] = acc[v];
  }
  __syncthreads();

  // ---- masks + duplicate-key count correction ----
  for (int c = tid; c < 128 * 256; c += 256) {
    const int i = c >> 8, j = c & 255;
    float s = SS[i * SS_STR + j] * kSCALE;
    const unsigned ti = qid[i], tj = kid[j];
    if (!kms[j])           s = kNEG_BIG;     // padding mask
    if (qhs[i] != khs[j])  s = kNEG_BIG;     // same-hash mask
    if (ti < tj)           s = kNEG_BIG;     // causal
    if (ti == tj)          s = kNEG_SMALL;   // self
    const unsigned cq = qck[i], ck = kck[j];
    int cnt = 0;
#pragma unroll
    for (int rr = 0; rr < kR; ++rr) {
      const unsigned a = (cq >> (4 * rr)) & 15u, c2 = (ck >> (4 * rr)) & 15u;
      cnt += (c2 == a) || (c2 == ((a + 15u) & 15u));
    }
    SS[i * SS_STR + j] = s - __logf((float)cnt);   // cnt in [1,4]
  }
  __syncthreads();

  // ---- per-row max/sumexp; in-place f32 -> bf16 P conversion ----
  if (tid < 128) {
    const float* f = SS + tid * SS_STR;
    float m = -3.0e38f;
    for (int j = 0; j < 256; ++j) m = fmaxf(m, f[j]);
    __bf16* prow = PP + tid * PP_STR;        // same bytes, write<=read offset
    float l = 0.f;
    for (int j = 0; j < 256; ++j) { const float e = __expf(f[j] - m); l += e; prow[j] = (__bf16)e; }
    const size_t idx = base + n * 128 + tid;
    Mg[idx] = m; Lg[idx] = l;
  }
  __syncthreads();

  // ---- O = P . V  (bf16 WMMA, K-dim 256 = 8x32), per-round unnormalized ----
  for (int dt = 0; dt < 8; ++dt) {
    v8f acc = {0.f, 0.f, 0.f, 0.f, 0.f, 0.f, 0.f, 0.f};
#pragma unroll
    for (int kb = 0; kb < 8; ++kb) {
      const int j0 = kb * 32;
      const __bf16* prow = PP + (i0 + lr) * PP_STR;
      const v16bf A = ldfrag(prow + j0 + lh * 8, prow + j0 + 16 + lh * 8);
      const __bf16* vrow = VT + (dt * 16 + lr) * VT_STR;
      const v16bf B = ldfrag(vrow + j0 + lh * 8, vrow + j0 + 16 + lh * 8);
      acc = __builtin_amdgcn_wmma_f32_16x16x32_bf16(false, A, false, B,
                                                    (short)0, acc, false, false);
    }
#pragma unroll
    for (int v = 0; v < 8; ++v)
      Og[(base + n * 128 + (i0 + v + lh * 8)) * kD + dt * 16 + lr] = acc[v];
  }
}

// ---------------------------------------------------------------- kernel 4 --
__global__ void lsh_combine(const int* __restrict__ oi, const float* __restrict__ Mg,
                            const float* __restrict__ Lg, const float* __restrict__ Og,
                            float* __restrict__ out) {
  const int t = blockIdx.x, b = blockIdx.y, d = threadIdx.x;
  float m[kR], l[kR]; int p[kR];
#pragma unroll
  for (int r = 0; r < kR; ++r) {
    const size_t base = ((size_t)b * kR + r) * kLEN;
    p[r] = oi[base + t]; m[r] = Mg[base + p[r]]; l[r] = Lg[base + p[r]];
  }
  float M = m[0];
#pragma unroll
  for (int r = 1; r < kR; ++r) M = fmaxf(M, m[r]);
  float Lt = 0.f;
#pragma unroll
  for (int r = 0; r < kR; ++r) Lt += l[r] * __expf(m[r] - M);
  const float inv = 1.0f / Lt;
  float acc = 0.f;
#pragma unroll
  for (int r = 0; r < kR; ++r) {
    const size_t base = ((size_t)b * kR + r) * kLEN;
    acc += __expf(m[r] - M) * inv * Og[(base + p[r]) * kD + d];
  }
  out[((size_t)b * kLEN + t) * kD + d] = acc;
}

// ------------------------------------------------------------------- launch --
extern "C" void kernel_launch(void* const* d_in, const int* in_sizes, int n_in,
                              void* d_out, int out_size, void* d_ws, size_t ws_size,
                              hipStream_t stream) {
  (void)in_sizes; (void)n_in; (void)out_size; (void)ws_size;
  const float*         query = (const float*)d_in[0];
  const float*         value = (const float*)d_in[1];
  const unsigned char* maskb = (const unsigned char*)d_in[2];
  const float*         randm = (const float*)d_in[3];

  char* ws = (char*)d_ws;
  __bf16*   fqb  = (__bf16*)  (ws + OFF_FQB);
  __bf16*   vb   = (__bf16*)  (ws + OFF_VB);
  float*    rmn  = (float*)   (ws + OFF_RMN);
  int*      hsh  = (int*)     (ws + OFF_HSH);
  int*      hi   = (int*)     (ws + OFF_HI);
  int*      sh   = (int*)     (ws + OFF_SH);
  int*      oi   = (int*)     (ws + OFF_OI);
  unsigned* chk  = (unsigned*)(ws + OFF_CHK);
  float*    Mg   = (float*)   (ws + OFF_M);
  float*    Lg   = (float*)   (ws + OFF_L);
  float*    Og   = (float*)   (ws + OFF_OG);
  float*    fout = (float*)d_out;

  (void)hipFuncSetAttribute((const void*)lsh_attn,
                            hipFuncAttributeMaxDynamicSharedMemorySize,
                            (int)SMEM3);

  lsh_rmnorm <<<dim3(kNB2, kR, kB),      128, 0, stream>>>(randm, rmn);
  lsh_hash   <<<dim3(kLEN, kB),          128, 0, stream>>>(query, value, rmn, fqb, vb, hsh);
  lsh_sort   <<<dim3(kR, kB),           1024, 0, stream>>>(hsh, hi, sh, oi);
  lsh_chunks <<<dim3(kB * kLEN / 256),   256, 0, stream>>>(oi, chk);
  lsh_attn   <<<dim3(kNB2, kR, kB), 256, SMEM3, stream>>>(fqb, vb, hi, sh, chk, maskb,
                                                          Mg, Lg, Og);
  lsh_combine<<<dim3(kLEN, kB),          128, 0, stream>>>(oi, Mg, Lg, Og, fout);
}
